// MoETransformer_10204842295375
// MI455X (gfx1250) — compile-verified, matrix operand order checked by reading, not run on hardware
//
#include <hip/hip_runtime.h>
#include <stdint.h>

// ---------------- problem constants ----------------
#define BB    4
#define SS    512
#define DDIM  1024
#define HH    16
#define VV    32000
#define EE    8
#define LL    2
#define FFDIM 4096
#define HDIM  64
#define NTOK  (BB * SS)   // 2048 tokens

// ---------------- vector types ----------------
typedef __attribute__((ext_vector_type(16))) __bf16 v16bf;
typedef __attribute__((ext_vector_type(8)))  float  v8f;
typedef __attribute__((ext_vector_type(4)))  unsigned int u32x4;
typedef __attribute__((ext_vector_type(4)))  float  f32x4;
typedef __attribute__((ext_vector_type(8)))  int    i32x8;
typedef __attribute__((ext_vector_type(4)))  int    i32x4;

union FragBF16 { v16bf v; u32x4 q[2]; };

// ---------------- CDNA5 feature detection (compile-safe guards) ----------
#if defined(__has_builtin)
#  if __has_builtin(__builtin_amdgcn_tensor_load_to_lds) && __has_builtin(__builtin_amdgcn_s_wait_tensorcnt)
#    define HAVE_TDM 1
#  endif
#  if __has_builtin(__builtin_amdgcn_global_load_async_to_lds_b128)
#    define HAVE_ASYNC 1
#  endif
#endif
#ifndef HAVE_TDM
#define HAVE_TDM 0
#endif
#ifndef HAVE_ASYNC
#define HAVE_ASYNC 0
#endif

#if HAVE_ASYNC
typedef const __attribute__((address_space(1))) void as1_cvoid;
typedef __attribute__((address_space(3)))       void as3_void;
#endif

__device__ __forceinline__ unsigned short f2bf(float f) {
  union { float f; unsigned int u; } c; c.f = f;
  unsigned int u = c.u;
  unsigned int r = u + 0x7FFFu + ((u >> 16) & 1u);   // round-to-nearest-even
  return (unsigned short)(r >> 16);
}

// ---------------- GEMM tiling ----------------
#define BLK_M 128
#define BLK_N 64
#define BLK_K 32
#define EP_F32        0
#define EP_RELU_BF16  1
#define EP_SCALED_ACC 2

// Stage the 128x32 bf16 A tile into LDS.
//  - TDM path: wave 0 builds a D# (2D tile: dim0=32 elems of 2B, dim1=128 rows,
//    dim0_stride=lda) and issues one tensor_load_to_lds; waits on TENSORcnt.
//  - async path: per-lane global_load_async_to_lds_b128; waits on ASYNCcnt.
//  - fallback: plain b128 copies through VGPRs.
__device__ __forceinline__ void stage_A_tile(const unsigned short* __restrict__ A,
                                             int lda, int bm, int k0,
                                             unsigned short* lA, int tid, int wid) {
#if HAVE_TDM
  if (wid == 0) {
    const unsigned short* gsrc = A + (size_t)bm * lda + k0;
    unsigned long long ga = (unsigned long long)(uintptr_t)gsrc;
    unsigned lds = (unsigned)(uintptr_t)lA;          // low 32 bits = DS offset
    u32x4 g0;
    g0[0] = 1u;                                      // count=1, user mode
    g0[1] = lds;                                     // lds_addr
    g0[2] = (unsigned)(ga & 0xFFFFFFFFu);            // global_addr[31:0]
    g0[3] = (unsigned)((ga >> 32) & 0x01FFFFFFu) | 0x80000000u; // [56:32] | type=2
    const unsigned td0 = (unsigned)lda;              // tensor_dim0
    const unsigned td1 = (unsigned)NTOK;             // tensor_dim1 (rows)
    const unsigned long long st0 = (unsigned long long)lda; // dim0 stride (elems)
    i32x8 g1;
    g1[0] = (int)(1u << 16);                                   // data_size=2B
    g1[1] = (int)((td0 & 0xFFFFu) << 16);                      // td0[15:0]
    g1[2] = (int)(((td0 >> 16) & 0xFFFFu) | ((td1 & 0xFFFFu) << 16));
    g1[3] = (int)(((td1 >> 16) & 0xFFFFu) | ((unsigned)BLK_K << 16)); // tile_dim0=32
    g1[4] = (int)(unsigned)BLK_M;                              // tile_dim1=128
    g1[5] = (int)(unsigned)(st0 & 0xFFFFFFFFu);
    g1[6] = (int)(unsigned)((st0 >> 32) & 0xFFFFu);
    g1[7] = 0;
    i32x4 z4; z4[0] = 0; z4[1] = 0; z4[2] = 0; z4[3] = 0;
    i32x8 z8;
    #pragma unroll
    for (int t = 0; t < 8; t++) z8[t] = 0;
    __builtin_amdgcn_tensor_load_to_lds(g0, g1, z4, z4, z8, 0);
    __builtin_amdgcn_s_wait_tensorcnt(0);
  }
  (void)tid;
#else
  const int row = tid >> 1;
  const int c16 = (tid & 1) * 16;
  const unsigned short* gsrc = A + (size_t)(bm + row) * lda + k0 + c16;
  unsigned short* ldst = &lA[row * BLK_K + c16];
  (void)wid;
#if HAVE_ASYNC
  __builtin_amdgcn_global_load_async_to_lds_b128(
      (as1_cvoid*)(uintptr_t)gsrc, (as3_void*)(uintptr_t)ldst, 0, 0);
  __builtin_amdgcn_global_load_async_to_lds_b128(
      (as1_cvoid*)(uintptr_t)gsrc, (as3_void*)(uintptr_t)ldst, 16, 0);
#if __has_builtin(__builtin_amdgcn_s_wait_asynccnt)
  __builtin_amdgcn_s_wait_asynccnt(0);
#else
  asm volatile("s_wait_asynccnt 0" ::: "memory");
#endif
#else
  const u32x4* s4 = (const u32x4*)gsrc;
  *(u32x4*)ldst       = s4[0];
  *(u32x4*)(ldst + 8) = s4[1];
#endif
#endif
}

// ---------------- generic WMMA bf16 GEMM ----------------
// C[M,N] = EP( A_bf16[M,K] * B + bias )
//   BKN=0 -> B fp32 [N,K] (out-major);  BKN=1 -> B fp32 [K,N] (transposed in LDS)
//   EP: 0 = Cf = x+bias ; 1 = Cb = bf16(relu(x+bias)) ; 2 = Cf += scale[row]*(x+bias)
template <int BKN, int EP>
__global__ __launch_bounds__(256)
void gemm_wmma_bf16(const unsigned short* __restrict__ A, int lda,
                    const float* __restrict__ B, int ldb,
                    const float* __restrict__ bias,
                    float* __restrict__ Cf, unsigned short* __restrict__ Cb, int ldc,
                    const float* __restrict__ scale, int scaleStride, int Kdim)
{
  __shared__ unsigned short lA[BLK_M * BLK_K];   // 8 KB bf16
  __shared__ unsigned short lB[BLK_N * BLK_K];   // 4 KB bf16, stored [n][k]

  const int tid  = threadIdx.x;
  const int bm   = blockIdx.y * BLK_M;
  const int bn   = blockIdx.x * BLK_N;
  const int lane = tid & 31;
  const int wid  = tid >> 5;
  const int wm   = (wid & 3) * 32;
  const int wn   = (wid >> 2) * 32;
  const int half = lane >> 4;
  const int r    = lane & 15;

  v8f acc[2][2];
  #pragma unroll
  for (int i = 0; i < 2; i++)
    #pragma unroll
    for (int j = 0; j < 2; j++)
      #pragma unroll
      for (int e = 0; e < 8; e++) acc[i][j][e] = 0.0f;

  for (int k0 = 0; k0 < Kdim; k0 += BLK_K) {
    __syncthreads();
    stage_A_tile(A, lda, bm, k0, lA, tid, wid);
    if (BKN == 0) {
      // B is [N,K]: 64 out-rows x 32 k = 128 chunks of 16 floats
      if (tid < 128) {
        const int row = tid >> 1;
        const int c16 = (tid & 1) * 16;
        const f32x4* src = (const f32x4*)(B + (size_t)(bn + row) * ldb + k0 + c16);
        f32x4 s0 = src[0], s1 = src[1], s2 = src[2], s3 = src[3];
        unsigned short* dst = &lB[row * BLK_K + c16];
        #pragma unroll
        for (int t = 0; t < 4; t++) dst[t]      = f2bf(s0[t]);
        #pragma unroll
        for (int t = 0; t < 4; t++) dst[4 + t]  = f2bf(s1[t]);
        #pragma unroll
        for (int t = 0; t < 4; t++) dst[8 + t]  = f2bf(s2[t]);
        #pragma unroll
        for (int t = 0; t < 4; t++) dst[12 + t] = f2bf(s3[t]);
        if (k0 + BLK_K < Kdim)
          __builtin_prefetch(B + (size_t)(bn + row) * ldb + k0 + BLK_K + c16, 0, 1);
      }
    } else {
      // B is [K,N]: 32 k-rows x 64 cols; transpose while storing to LDS
      const int k  = tid >> 3;
      const int c8 = (tid & 7) * 8;
      const f32x4* src = (const f32x4*)(B + (size_t)(k0 + k) * ldb + bn + c8);
      f32x4 s0 = src[0], s1 = src[1];
      #pragma unroll
      for (int t = 0; t < 4; t++) lB[(c8 + t) * BLK_K + k]     = f2bf(s0[t]);
      #pragma unroll
      for (int t = 0; t < 4; t++) lB[(c8 + 4 + t) * BLK_K + k] = f2bf(s1[t]);
      if (k0 + BLK_K < Kdim)
        __builtin_prefetch(B + (size_t)(k0 + k + BLK_K) * ldb + bn + c8, 0, 1);
    }
    __syncthreads();

    // per-lane fragment loads per documented 16-bit WMMA VGPR layouts
    FragBF16 afr[2], bfr[2];
    #pragma unroll
    for (int i = 0; i < 2; i++) {
      const int row = wm + i * 16 + r;
      afr[i].q[0] = *(const u32x4*)&lA[row * BLK_K + half * 8];
      afr[i].q[1] = *(const u32x4*)&lA[row * BLK_K + 16 + half * 8];
    }
    #pragma unroll
    for (int j = 0; j < 2; j++) {
      const int col = wn + j * 16 + r;
      bfr[j].q[0] = *(const u32x4*)&lB[col * BLK_K + half * 8];
      bfr[j].q[1] = *(const u32x4*)&lB[col * BLK_K + 16 + half * 8];
    }
    #pragma unroll
    for (int i = 0; i < 2; i++)
      #pragma unroll
      for (int j = 0; j < 2; j++)
        acc[i][j] = __builtin_amdgcn_wmma_f32_16x16x32_bf16(
            false, afr[i].v, false, bfr[j].v, (short)0, acc[i][j], false, false);
  }

  // epilogue: C layout -> lane: col = r, row = 8*half + v (per 16x16 tile)
  #pragma unroll
  for (int i = 0; i < 2; i++) {
    #pragma unroll
    for (int j = 0; j < 2; j++) {
      const int col  = bn + wn + j * 16 + r;
      const int row0 = bm + wm + i * 16 + half * 8;
      const float bv = bias ? bias[col] : 0.0f;
      #pragma unroll
      for (int vi = 0; vi < 8; vi++) {
        const int row  = row0 + vi;
        const float x  = acc[i][j][vi] + bv;
        const size_t idx = (size_t)row * ldc + col;
        if (EP == EP_F32) {
          Cf[idx] = x;
        } else if (EP == EP_RELU_BF16) {
          Cb[idx] = f2bf(x > 0.0f ? x : 0.0f);
        } else {
          Cf[idx] += scale[(size_t)row * scaleStride] * x;
        }
      }
    }
  }
}

// ---------------- fp32 -> bf16 conversion ----------------
__global__ void cvt_bf16_kernel(const float* __restrict__ src,
                                unsigned short* __restrict__ dst, int n) {
  int i = blockIdx.x * blockDim.x + threadIdx.x;
  if (i < n) dst[i] = f2bf(src[i]);
}

// ---------------- embedding + positional ----------------
__global__ void embed_kernel(const int* __restrict__ x, const float* __restrict__ emb,
                             const float* __restrict__ pos, float* __restrict__ h) {
  size_t i = (size_t)blockIdx.x * blockDim.x + threadIdx.x;
  if (i >= (size_t)NTOK * DDIM) return;
  int n = (int)(i / DDIM), d = (int)(i % DDIM);
  int s = n % SS;
  h[i] = emb[(size_t)x[n] * DDIM + d] + pos[(size_t)s * DDIM + d];
}

// ---------------- batch-axis attention (reference bug: softmax over B=4) ----
__global__ void attn_kernel(const float* __restrict__ qkv, float* __restrict__ o) {
  int t = blockIdx.x * blockDim.x + threadIdx.x;   // S*H*B = 32768
  if (t >= SS * HH * BB) return;
  const int i  = t & (BB - 1);
  const int hh = (t >> 2) & (HH - 1);
  const int s  = t >> 6;
  const float* q = qkv + (size_t)(i * SS + s) * 3 * DDIM + hh * HDIM;
  float p[BB];
  float mx = -1e30f;
  #pragma unroll
  for (int j = 0; j < BB; j++) {
    const float* kr = qkv + (size_t)(j * SS + s) * 3 * DDIM + DDIM + hh * HDIM;
    float d = 0.0f;
    for (int c = 0; c < HDIM; c++) d += q[c] * kr[c];
    p[j] = d * 0.125f;
    mx = fmaxf(mx, p[j]);
  }
  float sum = 0.0f;
  #pragma unroll
  for (int j = 0; j < BB; j++) { p[j] = __expf(p[j] - mx); sum += p[j]; }
  const float inv = 1.0f / sum;
  #pragma unroll
  for (int j = 0; j < BB; j++) p[j] *= inv;
  const float* v0 = qkv + (size_t)(0 * SS + s) * 3 * DDIM + 2 * DDIM + hh * HDIM;
  const float* v1 = qkv + (size_t)(1 * SS + s) * 3 * DDIM + 2 * DDIM + hh * HDIM;
  const float* v2 = qkv + (size_t)(2 * SS + s) * 3 * DDIM + 2 * DDIM + hh * HDIM;
  const float* v3 = qkv + (size_t)(3 * SS + s) * 3 * DDIM + 2 * DDIM + hh * HDIM;
  float* dst = o + (size_t)(i * SS + s) * DDIM + hh * HDIM;
  for (int c = 0; c < HDIM; c++)
    dst[c] = p[0] * v0[c] + p[1] * v1[c] + p[2] * v2[c] + p[3] * v3[c];
}

// ---------------- LayerNorm over D=1024, one block per token ----------------
__global__ __launch_bounds__(256)
void ln_kernel(const float* __restrict__ inA, const float* __restrict__ inB,
               const float* __restrict__ g, const float* __restrict__ b,
               float* __restrict__ out1, float* __restrict__ out2) {
  __shared__ float red[256];
  const int n = blockIdx.x, tid = threadIdx.x;
  const float* a  = inA + (size_t)n * DDIM;
  const float* a2 = inB ? inB + (size_t)n * DDIM : nullptr;
  float v[4], s = 0.0f;
  #pragma unroll
  for (int t = 0; t < 4; t++) {
    int d = tid + t * 256;
    float x = a[d] + (a2 ? a2[d] : 0.0f);
    v[t] = x; s += x;
  }
  red[tid] = s; __syncthreads();
  for (int st = 128; st > 0; st >>= 1) { if (tid < st) red[tid] += red[tid + st]; __syncthreads(); }
  const float mu = red[0] / DDIM;
  __syncthreads();
  float sq = 0.0f;
  #pragma unroll
  for (int t = 0; t < 4; t++) { float d = v[t] - mu; sq += d * d; }
  red[tid] = sq; __syncthreads();
  for (int st = 128; st > 0; st >>= 1) { if (tid < st) red[tid] += red[tid + st]; __syncthreads(); }
  const float inv = rsqrtf(red[0] / DDIM + 1e-5f);
  #pragma unroll
  for (int t = 0; t < 4; t++) {
    int d = tid + t * 256;
    float y = (v[t] - mu) * inv * g[d] + b[d];
    out1[(size_t)n * DDIM + d] = y;
    if (out2) out2[(size_t)n * DDIM + d] = y;
  }
}

// ---------------- gate: softmax, top-2 renorm, combine weights, load sums ----
__global__ void gate_kernel(const float* __restrict__ h, const float* __restrict__ gw,
                            const float* __restrict__ gb, float* __restrict__ comb,
                            float* __restrict__ loadsum) {
  int n = blockIdx.x * blockDim.x + threadIdx.x;
  if (n >= NTOK) return;
  const float* hr = h + (size_t)n * DDIM;
  float lg[EE];
  #pragma unroll
  for (int e = 0; e < EE; e++) lg[e] = gb[e];
  for (int d = 0; d < DDIM; d++) {
    float hv = hr[d];
    #pragma unroll
    for (int e = 0; e < EE; e++) lg[e] += hv * gw[d * EE + e];
  }
  float mx = lg[0];
  #pragma unroll
  for (int e = 1; e < EE; e++) mx = fmaxf(mx, lg[e]);
  float gp[EE], sum = 0.0f;
  #pragma unroll
  for (int e = 0; e < EE; e++) { gp[e] = __expf(lg[e] - mx); sum += gp[e]; }
  const float inv = 1.0f / sum;
  #pragma unroll
  for (int e = 0; e < EE; e++) { gp[e] *= inv; atomicAdd(&loadsum[e], gp[e]); }
  int i1 = 0;
  #pragma unroll
  for (int e = 1; e < EE; e++) if (gp[e] > gp[i1]) i1 = e;
  int i2 = (i1 == 0) ? 1 : 0;
  #pragma unroll
  for (int e = 0; e < EE; e++) if (e != i1 && gp[e] > gp[i2]) i2 = e;
  const float e2 = __expf(gp[i2] - gp[i1]);
  const float w1 = 1.0f / (1.0f + e2);
  const float w2 = e2 / (1.0f + e2);
  #pragma unroll
  for (int e = 0; e < EE; e++)
    comb[(size_t)n * EE + e] = (e == i1) ? w1 : (e == i2) ? w2 : 0.0f;
}

// ---------------- load-balance loss helpers ----------------
__global__ void zero_kernel(float* p, int n) {
  int i = blockIdx.x * blockDim.x + threadIdx.x;
  if (i < n) p[i] = 0.0f;
}
__global__ void loss_accum_kernel(const float* __restrict__ loadsum, float* __restrict__ loss) {
  if (threadIdx.x == 0 && blockIdx.x == 0) {
    float el[EE], m = 0.0f;
    for (int e = 0; e < EE; e++) { el[e] = loadsum[e] / (float)NTOK; m += el[e]; }
    m /= (float)EE;
    float var = 0.0f;
    for (int e = 0; e < EE; e++) { float d = el[e] - m; var += d * d; }
    var /= (float)(EE - 1);
    loss[0] += (float)EE * var;
  }
}
__global__ void write_loss_kernel(const float* __restrict__ loss, float* __restrict__ dst) {
  if (threadIdx.x == 0 && blockIdx.x == 0) dst[0] = loss[0];
}

// ---------------- host launcher ----------------
extern "C" void kernel_launch(void* const* d_in, const int* in_sizes, int n_in,
                              void* d_out, int out_size, void* d_ws, size_t ws_size,
                              hipStream_t stream) {
  (void)in_sizes; (void)n_in; (void)out_size;
  const int*   x      = (const int*)  d_in[0];
  const float* emb    = (const float*)d_in[1];
  const float* pos    = (const float*)d_in[2];
  const float* in_w   = (const float*)d_in[3];
  const float* in_b   = (const float*)d_in[4];
  const float* ow     = (const float*)d_in[5];
  const float* ob     = (const float*)d_in[6];
  const float* gate_w = (const float*)d_in[7];
  const float* gate_b = (const float*)d_in[8];
  const float* w1     = (const float*)d_in[9];
  const float* b1     = (const float*)d_in[10];
  const float* w2     = (const float*)d_in[11];
  const float* b2     = (const float*)d_in[12];
  const float* ln1g   = (const float*)d_in[13];
  const float* ln1b   = (const float*)d_in[14];
  const float* ln2g   = (const float*)d_in[15];
  const float* ln2b   = (const float*)d_in[16];
  const float* out_w  = (const float*)d_in[17];
  const float* out_b  = (const float*)d_in[18];
  float* logits = (float*)d_out;

  char* ws = (char*)d_ws;
  size_t off = 0;
  auto carve = [&](size_t bytes) { size_t o = off; off = (off + bytes + 255) & ~(size_t)255; return o; };
  float*          h    = (float*)(ws + carve((size_t)NTOK * DDIM * 4));
  float*          qkv  = (float*)(ws + carve((size_t)NTOK * 3 * DDIM * 4));
  float*          attn = (float*)(ws + carve((size_t)NTOK * DDIM * 4));
  float*          proj = (float*)(ws + carve((size_t)NTOK * DDIM * 4));
  float*          accb = (float*)(ws + carve((size_t)NTOK * DDIM * 4));
  unsigned short* actb = (unsigned short*)(ws + carve((size_t)NTOK * DDIM * 2));
  unsigned short* midb = (unsigned short*)(ws + carve((size_t)NTOK * FFDIM * 2));
  float*          comb = (float*)(ws + carve((size_t)NTOK * EE * 4));
  float*          lsum = (float*)(ws + carve(EE * 4));
  float*          loss = (float*)(ws + carve(4));
  if (off > ws_size) return;

  const dim3 blk(256);
  const int nAct = NTOK * DDIM;

  embed_kernel<<<((size_t)NTOK * DDIM + 255) / 256, blk, 0, stream>>>(x, emb, pos, h);
  zero_kernel<<<1, 32, 0, stream>>>(loss, 1);

  for (int l = 0; l < LL; l++) {
    // QKV: [2048,1024] x [3072,1024]^T
    cvt_bf16_kernel<<<nAct / 256, blk, 0, stream>>>(h, actb, nAct);
    gemm_wmma_bf16<0, EP_F32><<<dim3(3 * DDIM / BLK_N, NTOK / BLK_M), blk, 0, stream>>>(
        actb, DDIM, in_w + (size_t)l * 3 * DDIM * DDIM, DDIM,
        in_b + (size_t)l * 3 * DDIM, qkv, nullptr, 3 * DDIM, nullptr, 0, DDIM);
    attn_kernel<<<(SS * HH * BB + 255) / 256, blk, 0, stream>>>(qkv, attn);
    // out-proj: [2048,1024] x [1024,1024]^T
    cvt_bf16_kernel<<<nAct / 256, blk, 0, stream>>>(attn, actb, nAct);
    gemm_wmma_bf16<0, EP_F32><<<dim3(DDIM / BLK_N, NTOK / BLK_M), blk, 0, stream>>>(
        actb, DDIM, ow + (size_t)l * DDIM * DDIM, DDIM,
        ob + (size_t)l * DDIM, proj, nullptr, DDIM, nullptr, 0, DDIM);
    // h = LN(h + attn_out); accb = h (MoE residual base)
    ln_kernel<<<NTOK, blk, 0, stream>>>(h, proj, ln1g + (size_t)l * DDIM,
                                        ln1b + (size_t)l * DDIM, h, accb);
    zero_kernel<<<1, 32, 0, stream>>>(lsum, EE);
    gate_kernel<<<(NTOK + 255) / 256, blk, 0, stream>>>(
        h, gate_w + (size_t)l * DDIM * EE, gate_b + (size_t)l * EE, comb, lsum);
    loss_accum_kernel<<<1, 32, 0, stream>>>(lsum, loss);
    // dense MoE: every expert on every token, weighted accumulate into accb
    cvt_bf16_kernel<<<nAct / 256, blk, 0, stream>>>(h, actb, nAct);
    for (int e = 0; e < EE; e++) {
      const float* w1e = w1 + (size_t)(l * EE + e) * DDIM * FFDIM;   // [D,FF] = [K,N]
      const float* b1e = b1 + (size_t)(l * EE + e) * FFDIM;
      const float* w2e = w2 + (size_t)(l * EE + e) * FFDIM * DDIM;   // [FF,D] = [K,N]
      const float* b2e = b2 + (size_t)(l * EE + e) * DDIM;
      gemm_wmma_bf16<1, EP_RELU_BF16><<<dim3(FFDIM / BLK_N, NTOK / BLK_M), blk, 0, stream>>>(
          actb, DDIM, w1e, FFDIM, b1e, nullptr, midb, FFDIM, nullptr, 0, DDIM);
      gemm_wmma_bf16<1, EP_SCALED_ACC><<<dim3(DDIM / BLK_N, NTOK / BLK_M), blk, 0, stream>>>(
          midb, FFDIM, w2e, DDIM, b2e, accb, nullptr, DDIM, comb + e, EE, FFDIM);
    }
    // h = LN(accb)
    ln_kernel<<<NTOK, blk, 0, stream>>>(accb, nullptr, ln2g + (size_t)l * DDIM,
                                        ln2b + (size_t)l * DDIM, h, nullptr);
  }

  // logits: [2048,1024] x [1024,32000]
  cvt_bf16_kernel<<<nAct / 256, blk, 0, stream>>>(h, actb, nAct);
  gemm_wmma_bf16<1, EP_F32><<<dim3(VV / BLK_N, NTOK / BLK_M), blk, 0, stream>>>(
      actb, DDIM, out_w, VV, out_b, logits, nullptr, VV, nullptr, 0, DDIM);
  write_loss_kernel<<<1, 32, 0, stream>>>(loss, logits + (size_t)NTOK * VV);
}
